// Generator_51316269252974
// MI455X (gfx1250) — matrix operand order repaired and checked by reading, not compiled, not run on hardware
//
#include <hip/hip_runtime.h>
#include <math.h>

// ---------------------------------------------------------------------------
// StyleGAN2 generator forward for MI455X (gfx1250), wave32 + WMMA bf16.
// Convs = implicit GEMM on v_wmma_f32_16x16x32_bf16, channels-last bf16
// staging, 4-way N register blocking (16x64 strip per wave):
//   per 32-K chunk = 2 A loads + 8 B loads -> 4 WMMAs (A reuse).
// No integer division anywhere on the device (grid-based decode + pow2 masks).
// ---------------------------------------------------------------------------

typedef __attribute__((ext_vector_type(16))) __bf16 v16bf;
typedef __attribute__((ext_vector_type(8)))  __bf16 v8bf;
typedef __attribute__((ext_vector_type(8)))  float  v8f;

#define SQRT2F 1.41421356237309504880f

__device__ __forceinline__ v8f wmma_bf16(v16bf a, v16bf b, v8f c) {
  return __builtin_amdgcn_wmma_f32_16x16x32_bf16(false, a, false, b,
                                                 (short)0, c, false, false);
}

// ---------------------------------------------------------------------------
// Tiny GEMV-style helpers (batch=2 -> VALU). grid = (cdiv(Out,128), B).
// ---------------------------------------------------------------------------

__global__ void pixelnorm_kernel(const float* __restrict__ z,
                                 float* __restrict__ out, int D) {
  __shared__ float red[512];
  int b = blockIdx.x, t = threadIdx.x;
  float v = z[b * D + t];
  red[t] = v * v;
  __syncthreads();
  for (int s = 256; s > 0; s >>= 1) {
    if (t < s) red[t] += red[t + s];
    __syncthreads();
  }
  out[b * D + t] = v * rsqrtf(red[0] / (float)D + 1e-8f);
}

__global__ void linear_kernel(const float* __restrict__ in,
                              const float* __restrict__ w,
                              const float* __restrict__ bias,
                              float* __restrict__ out,
                              int In, int Out,
                              float wscale, float bmul, int act) {
  int o = blockIdx.x * blockDim.x + threadIdx.x;
  int b = blockIdx.y;
  if (o >= Out) return;
  const float* ir = in + (size_t)b * In;
  const float* wr = w + (size_t)o * In;
  float acc = 0.f;
  for (int j = 0; j < In; ++j) acc += ir[j] * wr[j];
  acc = acc * wscale + bias[o] * bmul;
  if (act) acc = (acc < 0.f ? 0.2f * acc : acc) * SQRT2F;
  out[(size_t)b * Out + o] = acc;
}

// grid = (cdiv(Cout,128), B)
__global__ void demod_kernel(const float* __restrict__ wgt,
                             const float* __restrict__ sty,
                             float* __restrict__ dc,
                             int Cin, int Cout, int kk2, float wscale) {
  int o = blockIdx.x * blockDim.x + threadIdx.x;
  int b = blockIdx.y;
  if (o >= Cout) return;
  const float* wr = wgt + (size_t)o * Cin * kk2;
  const float* sr = sty + (size_t)b * Cin;
  float acc = 0.f;
  for (int ci = 0; ci < Cin; ++ci) {
    float ws = 0.f;
    for (int t = 0; t < kk2; ++t) {
      float wv = wr[ci * kk2 + t] * wscale;
      ws += wv * wv;
    }
    float sv = sr[ci];
    acc += ws * sv * sv;
  }
  dc[(size_t)b * Cout + o] = rsqrtf(acc + 1e-8f);
}

// ---------------------------------------------------------------------------
// Staging passes
// ---------------------------------------------------------------------------

__global__ void zero_kernel(unsigned int* __restrict__ p, int n) {
  int i = blockIdx.x * blockDim.x + threadIdx.x;
  if (i < n) p[i] = 0u;
}

// NCHW f32 -> NHWC bf16 with fused style modulation (LDS-tiled transpose).
// HW and Cin always multiples of 16 -> no bounds checks.
__global__ void __launch_bounds__(256)
premod_kernel(const float* __restrict__ x, const float* __restrict__ sty,
              __bf16* __restrict__ out, int Cin, int HW, int bcast) {
  __shared__ float tile[16][17];
  const int p0 = blockIdx.x * 16, c0 = blockIdx.y * 16, b = blockIdx.z;
  const int tx = threadIdx.x, ty = threadIdx.y;
  const float* src = x + (size_t)(bcast ? 0 : b) * Cin * HW;
  int c = c0 + ty;
  tile[ty][tx] = src[(size_t)c * HW + p0 + tx] * sty[(size_t)b * Cin + c];
  __syncthreads();
  out[((size_t)b * HW + p0 + ty) * Cin + c0 + tx] = (__bf16)tile[tx][ty];
}

// Weights f32 [Cout,Cin,ks,ks] -> bf16 [Cout][cell][Cin], scaled, opt flip.
// grid = (cdiv(Cin,256), kk2, Cout)
__global__ void prep_weight_kernel(const float* __restrict__ w,
                                   __bf16* __restrict__ out,
                                   int Cin, int kk2, int flip, float wscale) {
  int ci = blockIdx.x * blockDim.x + threadIdx.x;
  int cell = blockIdx.y;
  int m = blockIdx.z;
  if (ci >= Cin) return;
  int scell = flip ? (kk2 - 1 - cell) : cell;
  out[((size_t)m * kk2 + cell) * Cin + ci] =
      (__bf16)(w[((size_t)m * Cin + ci) * kk2 + scell] * wscale);
}

// ---------------------------------------------------------------------------
// WMMA implicit-GEMM conv (stride 1, pad ks/2), 4-way N register blocking.
// xm: [B,H,W,Cin] bf16 (pre-modulated), wb: [Cout][cell][Cin] bf16 (scaled).
// grid = (cdiv(W,256), H, B<<mShift); 4 waves/block; wave strip = 16x64.
// mode 0: to_rgb (bias only).  mode 1: styled (demod + noise + lrelu*sqrt2).
// ---------------------------------------------------------------------------
__global__ void __launch_bounds__(128)
conv_wmma_kernel(const __bf16* __restrict__ xm, const __bf16* __restrict__ wb,
                 const __bf16* __restrict__ zpage,
                 const float* __restrict__ dc, const float* __restrict__ noise,
                 const float* __restrict__ nwp, const float* __restrict__ bias,
                 float* __restrict__ out,
                 int Cin, int Cout, int H, int W, int ks, int mode,
                 int mShift) {
  const int lane = threadIdx.x & 31;
  const int wv   = threadIdx.x >> 5;
  const int col  = lane & 15;
  const int grp  = lane >> 4;
  const int xBase = (blockIdx.x * 4 + wv) * 64;
  if (xBase >= W) return;                 // wave-uniform
  const int y  = blockIdx.y;
  const int mt = blockIdx.z & ((1 << mShift) - 1);
  const int b  = blockIdx.z >> mShift;
  const int mBase = mt * 16;
  const int pad = ks >> 1;
  const int kk2 = ks * ks;
  const int K = kk2 * Cin;

  int  xs[4];
  bool xok[4];
#pragma unroll
  for (int t = 0; t < 4; ++t) {
    xs[t] = xBase + t * 16 + col;
    xok[t] = xs[t] < W;
  }
  const int m = mBase + col;
  const __bf16* wrow = (m < Cout) ? (wb + (size_t)m * K) : zpage;
  const bool mZ = (m >= Cout);

  v8f acc[4];
#pragma unroll
  for (int t = 0; t < 4; ++t) acc[t] = (v8f){0.f,0.f,0.f,0.f,0.f,0.f,0.f,0.f};

  int cell = 0;
  for (int ky = 0; ky < ks; ++ky) {
    const int yy = y + ky - pad;
    if (yy < 0 || yy >= H) { cell += ks; continue; }   // uniform skip
    const __bf16* xrowb = xm + ((size_t)b * H + yy) * W * Cin;
    for (int kx = 0; kx < ks; ++kx, ++cell) {
      const __bf16* bp[4];
#pragma unroll
      for (int t = 0; t < 4; ++t) {
        const int xx = xs[t] + kx - pad;
        const bool v = xok[t] && xx >= 0 && xx < W;
        bp[t] = v ? (xrowb + (size_t)xx * Cin) : zpage;
      }
      const __bf16* ap = mZ ? zpage : (wrow + cell * Cin);
      for (int c = 0; c < Cin; c += 32) {
        v16bf av = *(const v16bf*)(ap + c + (grp << 4));
#pragma unroll
        for (int t = 0; t < 4; ++t) {
          v16bf bv = *(const v16bf*)(bp[t] + c + (grp << 4));
          acc[t] = wmma_bf16(av, bv, acc[t]);
        }
      }
    }
  }

  // epilogue: base pointer per lane; D layout: M = v + 8*grp, N = lane&15
  const int HW = H * W;
  float* outp = out + (((size_t)b * Cout + mBase + (grp << 3)) * H + y) * W;
  const float nw = (mode == 1) ? nwp[0] : 0.f;
  const int moB = mBase + (grp << 3);
#pragma unroll
  for (int t = 0; t < 4; ++t) {
    if (!xok[t]) continue;
    const int x = xs[t];
    const float nz =
        (mode == 1) ? noise[((size_t)b * H + y) * W + x] : 0.f;
#pragma unroll
    for (int v = 0; v < 8; ++v) {
      if (moB + v < Cout) {
        float o = acc[t][v];
        if (mode == 1) {
          o *= dc[(size_t)b * Cout + moB + v];
          o += nw * nz;
          o += bias[moB + v];
          o = (o < 0.f ? 0.2f * o : o) * SQRT2F;
        } else {
          o += bias[moB + v];
        }
        outp[(size_t)v * HW + x] = o;
      }
    }
  }
}

// ---------------------------------------------------------------------------
// WMMA transposed conv: stride-2, pad(2,2), kernel flip baked into wb,
// 4-way N register blocking. tc[B,Cout,Ht,Ht], Ht = 2H+1.
// grid = (cdiv(Ht,256), Ht, B<<mShift).
// ---------------------------------------------------------------------------
__global__ void __launch_bounds__(128)
tconv_wmma_kernel(const __bf16* __restrict__ xm, const __bf16* __restrict__ wb,
                  const __bf16* __restrict__ zpage, float* __restrict__ out,
                  int Cin, int Cout, int H, int Ht, int mShift) {
  const int lane = threadIdx.x & 31;
  const int wv   = threadIdx.x >> 5;
  const int col  = lane & 15;
  const int grp  = lane >> 4;
  const int xBase = (blockIdx.x * 4 + wv) * 64;
  if (xBase >= Ht) return;
  const int Y  = blockIdx.y;
  const int mt = blockIdx.z & ((1 << mShift) - 1);
  const int b  = blockIdx.z >> mShift;
  const int mBase = mt * 16;
  const int K = 9 * Cin;

  int  Xs[4];
  bool xok[4];
#pragma unroll
  for (int t = 0; t < 4; ++t) {
    Xs[t] = xBase + t * 16 + col;
    xok[t] = Xs[t] < Ht;
  }
  const int m = mBase + col;
  const __bf16* wrow = (m < Cout) ? (wb + (size_t)m * K) : zpage;
  const bool mZ = (m >= Cout);

  v8f acc[4];
#pragma unroll
  for (int t = 0; t < 4; ++t) acc[t] = (v8f){0.f,0.f,0.f,0.f,0.f,0.f,0.f,0.f};

  int cell = 0;
  for (int ky = 0; ky < 3; ++ky) {
    const int pyy = Y + ky - 2;
    if (pyy < 0 || (pyy & 1) || (pyy >> 1) >= H) { cell += 3; continue; }
    const int iy = pyy >> 1;
    const __bf16* xrowb = xm + ((size_t)b * H + iy) * H * Cin;
    for (int kx = 0; kx < 3; ++kx, ++cell) {
      const __bf16* bp[4];
#pragma unroll
      for (int t = 0; t < 4; ++t) {
        const int pxx = Xs[t] + kx - 2;
        const int ix = pxx >> 1;
        const bool v = xok[t] && pxx >= 0 && !(pxx & 1) && ix < H;
        bp[t] = v ? (xrowb + (size_t)ix * Cin) : zpage;
      }
      const __bf16* ap = mZ ? zpage : (wrow + cell * Cin);
      for (int c = 0; c < Cin; c += 32) {
        v16bf av = *(const v16bf*)(ap + c + (grp << 4));
#pragma unroll
        for (int t = 0; t < 4; ++t) {
          v16bf bv = *(const v16bf*)(bp[t] + c + (grp << 4));
          acc[t] = wmma_bf16(av, bv, acc[t]);
        }
      }
    }
  }

  const int HtHt = Ht * Ht;
  float* outp = out + (((size_t)b * Cout + mBase + (grp << 3)) * Ht + Y) * Ht;
  const int moB = mBase + (grp << 3);
#pragma unroll
  for (int t = 0; t < 4; ++t) {
    if (!xok[t]) continue;
#pragma unroll
    for (int v = 0; v < 8; ++v) {
      if (moB + v < Cout)
        outp[(size_t)v * HtHt + Xs[t]] = acc[t][v];
    }
  }
}

// ---------------------------------------------------------------------------
// 4x4 binomial FIR (gain 4, pad(1,1)) + demod + noise + bias + lrelu*sqrt2.
// grid = (cdiv(Ho*Ho,256), C, B); Ho is a power of two (decode via lg).
// ---------------------------------------------------------------------------
__global__ void blur_post_kernel(const float* __restrict__ tc,
                                 const float* __restrict__ dc,
                                 const float* __restrict__ noise,
                                 const float* __restrict__ nwp,
                                 const float* __restrict__ actb,
                                 float* __restrict__ out,
                                 int C, int Ht, int Ho, int lgHo) {
  const float K1[4] = {1.f, 3.f, 3.f, 1.f};
  const int p = blockIdx.x * blockDim.x + threadIdx.x;
  const int c = blockIdx.y, b = blockIdx.z;
  if (p >= (Ho << lgHo)) return;
  const int X = p & (Ho - 1);
  const int Y = p >> lgHo;
  const float* t = tc + ((size_t)b * C + c) * Ht * Ht;
  float acc = 0.f;
  for (int fy = 0; fy < 4; ++fy) {
    int ty = Y + fy - 1;
    if (ty < 0 || ty >= Ht) continue;
    for (int fx = 0; fx < 4; ++fx) {
      int tx = X + fx - 1;
      if (tx < 0 || tx >= Ht) continue;
      acc += K1[fy] * K1[fx] * 0.0625f * t[(size_t)ty * Ht + tx];
    }
  }
  float o = acc * dc[(size_t)b * C + c];
  o += nwp[0] * noise[((size_t)b << (2 * lgHo)) + p];
  o += actb[c];
  o = (o < 0.f ? 0.2f * o : o) * SQRT2F;
  out[(((size_t)b * C + c) << (2 * lgHo)) + p] = o;
}

// ---------------------------------------------------------------------------
// Skip path: zero-insert x2 + 4x4 FIR (gain 4), pad(2,2), ADD into out.
// grid = (cdiv(Ho*Ho,256), C, B).
// ---------------------------------------------------------------------------
__global__ void skip_up_add_kernel(const float* __restrict__ skip,
                                   float* __restrict__ out,
                                   int C, int H, int Ho, int lgHo) {
  const float K1[4] = {1.f, 3.f, 3.f, 1.f};
  const int p = blockIdx.x * blockDim.x + threadIdx.x;
  const int c = blockIdx.y, b = blockIdx.z;
  if (p >= (Ho << lgHo)) return;
  const int X = p & (Ho - 1);
  const int Y = p >> lgHo;
  const float* s = skip + ((size_t)b * C + c) * H * H;
  float acc = 0.f;
  for (int fy = 0; fy < 4; ++fy) {
    int pyy = Y + fy - 2;
    if (pyy < 0 || (pyy & 1)) continue;
    int iy = pyy >> 1;
    if (iy >= H) continue;
    for (int fx = 0; fx < 4; ++fx) {
      int pxx = X + fx - 2;
      if (pxx < 0 || (pxx & 1)) continue;
      int ix = pxx >> 1;
      if (ix >= H) continue;
      acc += K1[fy] * K1[fx] * 0.0625f * s[(size_t)iy * H + ix];
    }
  }
  out[(((size_t)b * C + c) << (2 * lgHo)) + p] += acc;
}

// ---------------------------------------------------------------------------
// Host orchestration
// ---------------------------------------------------------------------------
static inline int cdiv(int a, int b) { return (a + b - 1) / b; }
static inline int ilog2(int v) { int l = 0; while ((1 << l) < v) ++l; return l; }

extern "C" void kernel_launch(void* const* d_in, const int* in_sizes, int n_in,
                              void* d_out, int out_size, void* d_ws,
                              size_t ws_size, hipStream_t stream) {
  // Input layout (setup_inputs insertion order):
  //  [0] z | [1..16] mapping (mw,mb)x8 | [17] const_input
  //  [18..102] convs: 17 x (weight, mod_w, mod_b, noise_w, act_b)
  //  [103..138] to_rgbs: 9 x (weight, mod_w, mod_b, bias)
  //  [139..155] noises
  const float* z        = (const float*)d_in[0];
  const float* const_in = (const float*)d_in[17];

  float* ws = (float*)d_ws;
  size_t off = 0;
  auto alloc = [&](size_t nelem) -> float* {
    float* p = ws + off;
    off += (nelem + 255) & ~(size_t)255;
    return p;
  };
  const size_t SKIP_SZ = (size_t)2 * 3 * 1024 * 1024;
  const size_t ACT_SZ  = (size_t)2 * 32 * 1024 * 1024;
  const size_t TC_SZ   = (size_t)2 * 32 * 1025 * 1025;
  float* lat0  = alloc(1024);
  float* lat1  = alloc(1024);
  float* sbuf  = alloc(1024);
  float* dcb   = alloc(1024);
  float* skipA = alloc(SKIP_SZ);
  float* skipB = alloc(SKIP_SZ);
  float* actA  = alloc(ACT_SZ);
  float* actB  = alloc(ACT_SZ);
  float* tcb   = alloc(TC_SZ);
  __bf16* xmb  = (__bf16*)alloc(ACT_SZ / 2 + 256);           // NHWC bf16
  __bf16* wbb  = (__bf16*)alloc((size_t)512 * 4608 / 2 + 256);
  __bf16* zpg  = (__bf16*)alloc(4096);                        // 8192 bf16 zeros

  zero_kernel<<<cdiv(4096, 256), 256, 0, stream>>>((unsigned int*)zpg, 4096);

  // ---- mapping network -----------------------------------------------------
  pixelnorm_kernel<<<2, 512, 0, stream>>>(z, lat0, 512);
  float* cur = lat0;
  float* nxt = lat1;
  const float mscale = 0.01f / sqrtf(512.f);
  for (int i = 0; i < 8; ++i) {
    linear_kernel<<<dim3(4, 2), 128, 0, stream>>>(
        cur, (const float*)d_in[1 + 2 * i], (const float*)d_in[2 + 2 * i], nxt,
        512, 512, mscale, 0.01f, 1);
    float* t = cur; cur = nxt; nxt = t;
  }
  const float* lat = cur;
  const float sscale = 1.f / sqrtf(512.f);

  auto style = [&](const float* mw, const float* mb, int cin) {
    linear_kernel<<<dim3(cdiv(cin, 128), 2), 128, 0, stream>>>(
        lat, mw, mb, sbuf, 512, cin, sscale, 1.f, 0);
  };
  auto premod = [&](const float* xin, int cin, int H, int bcast) {
    dim3 blk(16, 16), grd(H * H / 16, cin / 16, 2);
    premod_kernel<<<grd, blk, 0, stream>>>(xin, sbuf, xmb, cin, H * H, bcast);
  };
  auto prepw = [&](const float* w, int cout, int cin, int kk2, int flip,
                   float wsc) {
    dim3 g(cdiv(cin, 256), kk2, cout);
    prep_weight_kernel<<<g, 256, 0, stream>>>(w, wbb, cin, kk2, flip, wsc);
  };
  auto demod = [&](const float* w, int cin, int cout, float wsc) {
    demod_kernel<<<dim3(cdiv(cout, 128), 2), 128, 0, stream>>>(
        w, sbuf, dcb, cin, cout, 9, wsc);
  };

  // ---- conv0 (styled, 4x4, 512->512, const input broadcast) ---------------
  {
    const float* w = (const float*)d_in[18];
    float wsc = 1.f / sqrtf(512.f * 9.f);
    style((const float*)d_in[19], (const float*)d_in[20], 512);
    premod(const_in, 512, 4, 1);
    prepw(w, 512, 512, 9, 0, wsc);
    demod(w, 512, 512, wsc);
    dim3 g(1, 4, 2 << 5);
    conv_wmma_kernel<<<g, 128, 0, stream>>>(
        xmb, wbb, zpg, dcb, (const float*)d_in[139], (const float*)d_in[21],
        (const float*)d_in[22], actB, 512, 512, 4, 4, 3, 1, 5);
  }
  // ---- to_rgb0 -> skipA ----------------------------------------------------
  {
    const float* w = (const float*)d_in[103];
    float wsc = 1.f / sqrtf(512.f);
    style((const float*)d_in[104], (const float*)d_in[105], 512);
    premod(actB, 512, 4, 0);
    prepw(w, 3, 512, 1, 0, wsc);
    dim3 g(1, 4, 2);
    conv_wmma_kernel<<<g, 128, 0, stream>>>(
        xmb, wbb, zpg, nullptr, nullptr, nullptr, (const float*)d_in[106],
        skipA, 512, 3, 4, 4, 1, 0, 0);
  }

  float* skipCur = skipA;
  static const int CHS[9] = {512, 512, 512, 512, 512, 256, 128, 64, 32};

  for (int lv = 0; lv < 8; ++lv) {
    int H = 4 << lv;      // input res
    int Ho = 2 * H;       // output res
    int Ht = 2 * H + 1;   // transposed-conv intermediate
    int lgHo = ilog2(Ho);
    int cin = CHS[lv], cout = CHS[lv + 1];
    int mS = ilog2(cout / 16);   // cout/16 is a power of two

    // ---- upsample styled conv (actB -> tcb -> blur -> actA) ----------------
    {
      int cb = 18 + 5 * (1 + 2 * lv);
      const float* w = (const float*)d_in[cb];
      float wsc = 1.f / sqrtf((float)cin * 9.f);
      style((const float*)d_in[cb + 1], (const float*)d_in[cb + 2], cin);
      premod(actB, cin, H, 0);
      prepw(w, cout, cin, 9, 1, wsc);   // flip baked in
      demod(w, cin, cout, wsc);
      dim3 g(cdiv(Ht, 256), Ht, 2 << mS);
      tconv_wmma_kernel<<<g, 128, 0, stream>>>(xmb, wbb, zpg, tcb, cin, cout,
                                               H, Ht, mS);
      dim3 gb(cdiv(Ho * Ho, 256), cout, 2);
      blur_post_kernel<<<gb, 256, 0, stream>>>(
          tcb, dcb, (const float*)d_in[139 + 1 + 2 * lv],
          (const float*)d_in[cb + 3], (const float*)d_in[cb + 4], actA,
          cout, Ht, Ho, lgHo);
    }
    // ---- same-res styled conv (actA -> actB) -------------------------------
    {
      int cb = 18 + 5 * (2 + 2 * lv);
      const float* w = (const float*)d_in[cb];
      float wsc = 1.f / sqrtf((float)cout * 9.f);
      style((const float*)d_in[cb + 1], (const float*)d_in[cb + 2], cout);
      premod(actA, cout, Ho, 0);
      prepw(w, cout, cout, 9, 0, wsc);
      demod(w, cout, cout, wsc);
      dim3 g(cdiv(Ho, 256), Ho, 2 << mS);
      conv_wmma_kernel<<<g, 128, 0, stream>>>(
          xmb, wbb, zpg, dcb, (const float*)d_in[139 + 2 + 2 * lv],
          (const float*)d_in[cb + 3], (const float*)d_in[cb + 4], actB,
          cout, cout, Ho, Ho, 3, 1, mS);
    }
    // ---- to_rgb + upsampled skip add ---------------------------------------
    {
      int rb = 103 + 4 * (lv + 1);
      const float* w = (const float*)d_in[rb];
      float wsc = 1.f / sqrtf((float)cout);
      style((const float*)d_in[rb + 1], (const float*)d_in[rb + 2], cout);
      premod(actB, cout, Ho, 0);
      prepw(w, 3, cout, 1, 0, wsc);
      float* tgt = (lv == 7) ? (float*)d_out : ((skipCur == skipA) ? skipB
                                                                   : skipA);
      dim3 g(cdiv(Ho, 256), Ho, 2);
      conv_wmma_kernel<<<g, 128, 0, stream>>>(
          xmb, wbb, zpg, nullptr, nullptr, nullptr, (const float*)d_in[rb + 3],
          tgt, cout, 3, Ho, Ho, 1, 0, 0);
      dim3 gs(cdiv(Ho * Ho, 256), 3, 2);
      skip_up_add_kernel<<<gs, 256, 0, stream>>>(skipCur, tgt, 3, H, Ho, lgHo);
      skipCur = tgt;
    }
  }
}
